// CombinedModel_30657476559309
// MI455X (gfx1250) — matrix operand order repaired and checked by reading, not compiled, not run on hardware
//
#include <hip/hip_runtime.h>
#include <hip/hip_bf16.h>

typedef __attribute__((ext_vector_type(16))) _Float16 v16h;
typedef __attribute__((ext_vector_type(8)))  _Float16 v8h;
typedef __attribute__((ext_vector_type(8)))  float    v8f;

#define LN_EPS 1e-5f

// ---------------------------------------------------------------------------
// Precompute: f32 -> f16 weight conversion
// ---------------------------------------------------------------------------
__global__ void convert_f16_kernel(const float* __restrict__ src,
                                   _Float16* __restrict__ dst, int n) {
  int i = blockIdx.x * blockDim.x + threadIdx.x;
  if (i < n) dst[i] = (_Float16)src[i];
}

// ---------------------------------------------------------------------------
// Precompute: M_p = w_o @ wv  (fold the two E x E projections; softmax==1)
// c_p = w_o @ bv + b_o
// grid = (256 rows n, 6 pairs), block = 64
// ---------------------------------------------------------------------------
struct PairPrepArgs {
  const float* wqkv[6];
  const float* bqkv[6];
  const float* wo[6];
  const float* bo[6];
  _Float16* mp;   // 6 * 256*256 halfs
  float* cp;      // 6 * 256 floats
};

__global__ void pair_prep_kernel(PairPrepArgs A) {
  const int p = blockIdx.y;
  const int n = blockIdx.x;
  const int t = threadIdx.x;                 // 0..63
  const float* wo_row = A.wo[p] + n * 256;
  const float* wv = A.wqkv[p] + 512 * 256;   // wv = rows [512,768) of w_qkv
  float a0 = 0.f, a1 = 0.f, a2 = 0.f, a3 = 0.f;
  for (int j = 0; j < 256; ++j) {
    float w = wo_row[j];
    const float* row = wv + j * 256;
    a0 += w * row[t];
    a1 += w * row[t + 64];
    a2 += w * row[t + 128];
    a3 += w * row[t + 192];
  }
  _Float16* dst = A.mp + p * 65536 + n * 256;
  dst[t]       = (_Float16)a0;
  dst[t + 64]  = (_Float16)a1;
  dst[t + 128] = (_Float16)a2;
  dst[t + 192] = (_Float16)a3;
  if (t == 0) {
    const float* bv = A.bqkv[p] + 512;
    float s = 0.f;
    for (int j = 0; j < 256; ++j) s += wo_row[j] * bv[j];
    A.cp[p * 256 + n] = s + A.bo[p][n];
  }
}

// ---------------------------------------------------------------------------
// Main fused kernel: 2 waves (64 threads) per block cooperating on 16 rows.
// Waves split the output-column (nt) range of every GEMM.
// LDS: 4 x f16[16][256] embeddings + f32[16][256] fused + f32[16][256] z = 64KB
// -> 4 blocks (8 waves) resident per WGP.
// ---------------------------------------------------------------------------
struct KArgs {
  const float *numeric, *label, *text, *dummy;
  const float *label_w, *label_b, *label_g, *label_bb;
  const float *num_w, *num_b, *num_g, *num_bb;
  const float *temp_b, *temp_g, *temp_bb;
  const float *final_b;
  const _Float16 *tw16, *fw16;
  const _Float16 *mp[6];
  const float *cp[6];
  const float *lng[6], *lnb[6];
  float* out;
};

// A-fragment (16x32 f16): lanes 0-15 row M=lane, halfs = K
// [32k+0..15]; lanes 16-31 same rows, K offset +16.
__device__ inline void load_afrags(const _Float16* __restrict__ A, int lane,
                                   v16h a[8]) {
  const int r = lane & 15;
  const int s = (lane >> 4) * 16;
#pragma unroll
  for (int kt = 0; kt < 8; ++kt) {
    const _Float16* p = A + r * 256 + kt * 32 + s;
    v8h lo = *(const v8h*)p;
    v8h hi = *(const v8h*)(p + 8);
    v16h v;
#pragma unroll
    for (int i = 0; i < 8; ++i) { v[i] = lo[i]; v[i + 8] = hi[i]; }
    a[kt] = v;
  }
}

// out = act(16x256) @ W^T for nt tiles [ntbase, ntbase+ntcnt).
// W stored row-major f16 (W[n][k]); B-fragment: lane holds column n=lane&15,
// contiguous 16 halfs of K.
__device__ inline void gemm_to_z(const v16h a[8], const _Float16* __restrict__ W,
                                 float* __restrict__ z, int lane,
                                 int ntbase, int ntcnt) {
  const int nl   = lane & 15;
  const int koff = (lane >> 4) * 16;
  const int mofs = (lane >> 4) * 8;
  for (int ni = 0; ni < ntcnt; ++ni) {
    const int nt = ntbase + ni;
    v8f acc = {};
    const _Float16* wp = W + (nt * 16 + nl) * 256 + koff;
#pragma unroll
    for (int kt = 0; kt < 8; ++kt) {
      v16h b = *(const v16h*)(wp + kt * 32);
      acc = __builtin_amdgcn_wmma_f32_16x16x32_f16(false, a[kt], false, b,
                                                   (short)0, acc, false, false);
    }
#pragma unroll
    for (int vg = 0; vg < 8; ++vg)
      z[(vg + mofs) * 256 + nt * 16 + nl] = acc[vg];  // D: m=vg(+8), n=lane&15
  }
}

// LayerNorm over a 16x256 f32 tile. Wave owns 8 rows: row = wave*8 + (lane>>2),
// 4 lanes per row (64 cols each). Row reduction = 2x shfl_xor (intra-wave).
// val = z + (q?) + (bias?); output = g*norm + b, either accumulated into
// `fused` (cross path) or written as f16 (embedding path).
template <bool ACC>
__device__ inline void ln_pass(const float* __restrict__ z,
                               const _Float16* __restrict__ q,
                               const float* __restrict__ bias,
                               const float* __restrict__ g,
                               const float* __restrict__ b,
                               _Float16* __restrict__ dst16,
                               float* __restrict__ fused,
                               int lane, int wave) {
  const int r  = wave * 8 + (lane >> 2);
  const int c0 = (lane & 3) * 64;
  float sum = 0.f, sq = 0.f;
  for (int i = 0; i < 64; ++i) {
    int c = c0 + i;
    float v = z[r * 256 + c];
    if (q)    v += (float)q[r * 256 + c];
    if (bias) v += bias[c];
    sum += v; sq += v * v;
  }
  sum += __shfl_xor(sum, 1); sum += __shfl_xor(sum, 2);
  sq  += __shfl_xor(sq, 1);  sq  += __shfl_xor(sq, 2);
  float mean = sum * (1.f / 256.f);
  float var  = sq * (1.f / 256.f) - mean * mean;
  float rstd = rsqrtf(var + LN_EPS);
  for (int i = 0; i < 64; ++i) {
    int c = c0 + i;
    float v = z[r * 256 + c];
    if (q)    v += (float)q[r * 256 + c];
    if (bias) v += bias[c];
    float o = g[c] * ((v - mean) * rstd) + b[c];
    if (ACC) fused[r * 256 + c] += o;
    else     dst16[r * 256 + c] = (_Float16)o;
  }
}

__global__ __launch_bounds__(64) void fused_model_kernel(KArgs P) {
  extern __shared__ __align__(16) char smem[];
  const int tid  = threadIdx.x;
  const int lane = tid & 31;
  const int wave = tid >> 5;
  const int rowbase = blockIdx.x * 16;

  _Float16* ebuf0 = (_Float16*)smem;             // text
  _Float16* ebuf1 = (_Float16*)(smem + 8192);    // label
  _Float16* ebuf2 = (_Float16*)(smem + 16384);   // numeric
  _Float16* ebuf3 = (_Float16*)(smem + 24576);   // temporal
  float* fused = (float*)(smem + 32768);
  float* zbuf  = (float*)(smem + 49152);
  _Float16* ebuf[4] = {ebuf0, ebuf1, ebuf2, ebuf3};

  // elementwise ownership: 8 rows per wave, quarter-row per lane
  const int r  = wave * 8 + (lane >> 2);
  const int c0 = (lane & 3) * 64;
  const int row = rowbase + r;

  // ---- text embedding (raw, no LN) + zero fused accumulator ----
  for (int i = 0; i < 64; ++i) {
    int c = c0 + i;
    ebuf0[r * 256 + c] = (_Float16)P.text[row * 256 + c];
    fused[r * 256 + c] = 0.f;
  }

  // ---- label embedding: LN(label @ W^T + b), K=9 (scalar VALU) ----
  {
    float lf[9];
#pragma unroll
    for (int k = 0; k < 9; ++k) lf[k] = P.label[row * 9 + k];
    for (int i = 0; i < 64; ++i) {
      int c = c0 + i;
      float s = 0.f;
#pragma unroll
      for (int k = 0; k < 9; ++k) s += lf[k] * P.label_w[c * 9 + k];
      zbuf[r * 256 + c] = s;
    }
    ln_pass<false>(zbuf, nullptr, P.label_b, P.label_g, P.label_bb, ebuf1,
                   nullptr, lane, wave);
  }

  // ---- numeric embedding: K=4 ----
  {
    float nf[4];
#pragma unroll
    for (int k = 0; k < 4; ++k) nf[k] = P.numeric[row * 4 + k];
    for (int i = 0; i < 64; ++i) {
      int c = c0 + i;
      float s = 0.f;
#pragma unroll
      for (int k = 0; k < 4; ++k) s += nf[k] * P.num_w[c * 4 + k];
      zbuf[r * 256 + c] = s;
    }
    ln_pass<false>(zbuf, nullptr, P.num_b, P.num_g, P.num_bb, ebuf2,
                   nullptr, lane, wave);
  }

  // ---- temporal embedding: LN(dummy @ W^T + b) via WMMA ----
  {
    for (int i = 0; i < 64; ++i) {
      int c = c0 + i;
      ebuf3[r * 256 + c] = (_Float16)P.dummy[row * 256 + c];
    }
    __syncthreads();                 // A-frags read all 16 rows
    v16h a[8];
    load_afrags(ebuf3, lane, a);
    gemm_to_z(a, P.tw16, zbuf, lane, wave * 8, 8);
    __syncthreads();                 // z columns produced by both waves
    ln_pass<false>(zbuf, nullptr, P.temp_b, P.temp_g, P.temp_bb, ebuf3,
                   nullptr, lane, wave);
  }

  // ---- 6 pairs x 2 directions: fused += LN(q + kv @ M_p^T + c_p) ----
  const int M1[6] = {0, 0, 0, 1, 1, 2};
  const int M2[6] = {1, 2, 3, 2, 3, 3};
  for (int p = 0; p < 6; ++p) {
    for (int d = 0; d < 2; ++d) {
      const int qi = d ? M2[p] : M1[p];
      const int ki = d ? M1[p] : M2[p];
      __syncthreads();               // emb writes visible; zbuf free to reuse
      v16h a[8];
      load_afrags(ebuf[ki], lane, a);
      gemm_to_z(a, P.mp[p], zbuf, lane, wave * 8, 8);
      __syncthreads();               // z columns produced by both waves
      ln_pass<true>(zbuf, ebuf[qi], P.cp[p], P.lng[p], P.lnb[p], nullptr,
                    fused, lane, wave);
    }
  }

  // ---- scale by 1/12, quantize to f16 (reuse text buffer; same-lane data) ----
  for (int i = 0; i < 64; ++i) {
    int c = c0 + i;
    ebuf0[r * 256 + c] = (_Float16)(fused[r * 256 + c] * (1.f / 12.f));
  }
  __syncthreads();                   // final A-frags read all 16 rows

  // ---- final: tanh(fused @ final_w^T + final_b), HID = 512 ----
  {
    v16h a[8];
    load_afrags(ebuf0, lane, a);
    const int nl   = lane & 15;
    const int koff = (lane >> 4) * 16;
    const int mofs = (lane >> 4) * 8;
    for (int ni = 0; ni < 16; ++ni) {
      const int nt = wave * 16 + ni;
      v8f acc = {};
      const _Float16* wp = P.fw16 + (nt * 16 + nl) * 256 + koff;
#pragma unroll
      for (int kt = 0; kt < 8; ++kt) {
        v16h b = *(const v16h*)(wp + kt * 32);
        acc = __builtin_amdgcn_wmma_f32_16x16x32_f16(false, a[kt], false, b,
                                                     (short)0, acc, false, false);
      }
      float fb = P.final_b[nt * 16 + nl];
#pragma unroll
      for (int vg = 0; vg < 8; ++vg) {
        P.out[(rowbase + vg + mofs) * 512 + nt * 16 + nl] = tanhf(acc[vg] + fb);
      }
    }
  }
}

// ---------------------------------------------------------------------------
// Host-side launcher
// ---------------------------------------------------------------------------
extern "C" void kernel_launch(void* const* d_in, const int* in_sizes, int n_in,
                              void* d_out, int out_size, void* d_ws, size_t ws_size,
                              hipStream_t stream) {
  (void)n_in; (void)out_size; (void)ws_size;
  // Workspace layout (bytes):
  //   [0,        131072)  temporal_w f16 (256x256)
  //   [131072,   393216)  final_w    f16 (512x256)
  //   [393216,  1179648)  M_p f16 x6 (256x256 each)
  //   [1179648, 1185792)  c_p f32 x6 (256 each)
  char* ws = (char*)d_ws;
  _Float16* tw16 = (_Float16*)(ws + 0);
  _Float16* fw16 = (_Float16*)(ws + 131072);
  _Float16* mp16 = (_Float16*)(ws + 393216);
  float*    cp   = (float*)   (ws + 1179648);

  convert_f16_kernel<<<256, 256, 0, stream>>>((const float*)d_in[12], tw16, 65536);
  convert_f16_kernel<<<512, 256, 0, stream>>>((const float*)d_in[16], fw16, 131072);

  PairPrepArgs pa;
  for (int p = 0; p < 6; ++p) {
    const int b = 18 + 6 * p;
    pa.wqkv[p] = (const float*)d_in[b + 0];
    pa.bqkv[p] = (const float*)d_in[b + 1];
    pa.wo[p]   = (const float*)d_in[b + 2];
    pa.bo[p]   = (const float*)d_in[b + 3];
  }
  pa.mp = mp16;
  pa.cp = cp;
  pair_prep_kernel<<<dim3(256, 6), 64, 0, stream>>>(pa);

  KArgs K;
  K.numeric = (const float*)d_in[0];
  K.label   = (const float*)d_in[1];
  K.text    = (const float*)d_in[2];
  K.dummy   = (const float*)d_in[3];
  K.label_w = (const float*)d_in[4];  K.label_b  = (const float*)d_in[5];
  K.label_g = (const float*)d_in[6];  K.label_bb = (const float*)d_in[7];
  K.num_w   = (const float*)d_in[8];  K.num_b    = (const float*)d_in[9];
  K.num_g   = (const float*)d_in[10]; K.num_bb   = (const float*)d_in[11];
  K.temp_b  = (const float*)d_in[13]; K.temp_g   = (const float*)d_in[14];
  K.temp_bb = (const float*)d_in[15];
  K.final_b = (const float*)d_in[17];
  K.tw16 = tw16;
  K.fw16 = fw16;
  for (int p = 0; p < 6; ++p) {
    const int b = 18 + 6 * p;
    K.mp[p]  = mp16 + p * 65536;
    K.cp[p]  = cp + p * 256;
    K.lng[p] = (const float*)d_in[b + 4];
    K.lnb[p] = (const float*)d_in[b + 5];
  }
  K.out = (float*)d_out;

  const int rows = in_sizes[2] / 256;  // N = 131072
  fused_model_kernel<<<rows / 16, 64, 65536, stream>>>(K);
}
